// Transolver_1236950582157
// MI455X (gfx1250) — matrix operand order repaired and checked by reading, not compiled, not run on hardware
//
#include <hip/hip_runtime.h>
#include <math.h>

#define BB 4
#define NN 16384
#define CIN_ 4
#define SD_ 3
#define WD 256
#define LL 8
#define MM 32
#define HH 8
#define FFD 1024

typedef __attribute__((ext_vector_type(16))) _Float16 v16h;
typedef __attribute__((ext_vector_type(8)))  _Float16 v8h;
typedef __attribute__((ext_vector_type(8)))  float    v8f;

union FragU { v16h v; v8h h[2]; };

// A fragment: 16x32 f16 tile, row-major with given stride (elements).
// ISA layout: lane row = lane&15; lane-half selects K subranges
// VGPR0-3 -> K = half*8 + 0..7 ; VGPR4-7 -> K = 16 + half*8 + 0..7
__device__ __forceinline__ v16h frag_a(const _Float16* base, int stride) {
  int l = threadIdx.x & 31;
  int row = l & 15, hf = l >> 4;
  const _Float16* r = base + row * stride + hf * 8;
  FragU u;
  u.h[0] = *(const v8h*)(r);
  u.h[1] = *(const v8h*)(r + 16);
  return u.v;
}

// B fragment: 32x16 f16 tile given B^T storage Bt[col][k] (stride = K-dim stride).
// lane half hf: elems 0-7 -> K = hf*8 + 0..7 ; elems 8-15 -> K = 16 + hf*8 + 0..7 (matches frag_a).
__device__ __forceinline__ v16h frag_b(const _Float16* bt, int stride) {
  int l = threadIdx.x & 31;
  int col = l & 15, hf = l >> 4;
  const _Float16* r = bt + col * stride + hf * 8;
  FragU u;
  u.h[0] = *(const v8h*)(r);
  u.h[1] = *(const v8h*)(r + 16);
  return u.v;
}

__device__ __forceinline__ v8f wmma16(v16h a, v16h b, v8f c) {
  return __builtin_amdgcn_wmma_f32_16x16x32_f16(false, a, false, b, (short)0, c,
                                                false, false);
}

// C/D layout: VGPR r -> m = r + 8*(lane>>4), n = lane&15
__device__ __forceinline__ int c_row(int r) { return r + 8 * ((threadIdx.x >> 4) & 1); }
__device__ __forceinline__ int c_col()      { return threadIdx.x & 15; }

__device__ __forceinline__ float wred_sum(float v) {
#pragma unroll
  for (int o = 16; o >= 1; o >>= 1) v += __shfl_xor(v, o, 32);
  return v;
}
__device__ __forceinline__ float wred_max(float v) {
#pragma unroll
  for (int o = 16; o >= 1; o >>= 1) v = fmaxf(v, __shfl_xor(v, o, 32));
  return v;
}

// ---------------- weight prep ----------------
__global__ void k_transpose_l(const float* __restrict__ src, _Float16* __restrict__ dst,
                              int R, int C, int Lc) {
  long i = (long)blockIdx.x * 256 + threadIdx.x;
  long tot = (long)Lc * R * C;
  if (i < tot) {
    int c = (int)(i % C);
    long lr = i / C;
    int r = (int)(lr % R);
    int ll = (int)(lr / R);
    dst[((long)ll * C + c) * R + r] = (_Float16)src[i];
  }
}
__global__ void k_prep_embed(const float* __restrict__ src, _Float16* __restrict__ dst) {
  // src [15][256] -> dst [256][32] zero padded K
  int i = blockIdx.x * 256 + threadIdx.x;
  if (i < 256 * 32) {
    int o = i >> 5, k = i & 31;
    dst[i] = (_Float16)((k < 15) ? src[k * 256 + o] : 0.f);
  }
}
__global__ void k_prep_proj(const float* __restrict__ src, _Float16* __restrict__ dst) {
  // src [256][4] -> dst [16][256] zero padded cols
  int i = blockIdx.x * 256 + threadIdx.x;
  if (i < 16 * 256) {
    int o = i >> 8, k = i & 255;
    dst[i] = (_Float16)((o < 4) ? src[k * 4 + o] : 0.f);
  }
}
__global__ void k_zero(float* __restrict__ Z, float* __restrict__ WS) {
  int i = blockIdx.x * 256 + threadIdx.x;
  if (i < BB * MM * WD) Z[i] = 0.f;
  if (i < BB * MM) WS[i] = 0.f;
}

// ---------------- embed ----------------
__global__ void __launch_bounds__(256) k_embed(
    const float* __restrict__ feat, const float* __restrict__ coords,
    const float* __restrict__ tnorm, const _Float16* __restrict__ embedWt,
    const float* __restrict__ embedB, float* __restrict__ X) {
  __shared__ _Float16 ldsA[32 * 32];
  long row0 = (long)blockIdx.x * 32;
  int b = (int)(row0 / NN);
  float t = tnorm[b] * 1000.f;
  for (int e = threadIdx.x; e < 32 * 32; e += 256) {
    int r = e >> 5, c = e & 31;
    long g = row0 + r;
    float val = 0.f;
    if (c < 4) val = feat[g * 4 + c];
    else if (c < 7) val = coords[g * 3 + (c - 4)];
    else if (c < 15) {
      int i = (c - 7) & 3;
      float omega = powf(1000.f, -(float)i * 0.25f);
      float ang = omega * t;
      val = (c < 11) ? sinf(ang) : cosf(ang);
    }
    ldsA[r * 32 + c] = (_Float16)val;
  }
  __syncthreads();
  int wv = threadIdx.x >> 5;
  int mt = wv & 1, ngb = (wv >> 1) * 4;
  v16h a = frag_a(ldsA + mt * 16 * 32, 32);
#pragma unroll
  for (int j = 0; j < 4; j++) {
    int nt = ngb + j;
    v16h bf = frag_b(embedWt + nt * 16 * 32, 32);
    v8f acc = {};
    acc = wmma16(a, bf, acc);
#pragma unroll
    for (int r = 0; r < 8; r++) {
      int mm = mt * 16 + c_row(r);
      int nn = nt * 16 + c_col();
      X[(row0 + mm) * WD + nn] = acc[r] + embedB[nn];
    }
  }
}

// ---------------- LN1 + slice softmax ----------------
__global__ void __launch_bounds__(256) k_slice(
    const float* __restrict__ X, const float* __restrict__ ln1g,
    const float* __restrict__ ln1b, const _Float16* __restrict__ sliceWt,
    const float* __restrict__ sliceB, _Float16* __restrict__ Hout,
    _Float16* __restrict__ Wout, float* __restrict__ WS, int layer) {
  __shared__ float ldsX[32 * 256];
  __shared__ _Float16 ldsH[32 * 256];
  __shared__ float ldsLg[32 * 32];
  __shared__ float ldsWS[32];
  long row0 = (long)blockIdx.x * 32;
  int b = (int)(row0 / NN);
  const float* gg = ln1g + layer * WD;
  const float* bb = ln1b + layer * WD;
  for (int e = threadIdx.x; e < 32 * 256; e += 256) ldsX[e] = X[row0 * WD + e];
  if (threadIdx.x < 32) ldsWS[threadIdx.x] = 0.f;
  __syncthreads();
  int wv = threadIdx.x >> 5, l = threadIdx.x & 31;
#pragma unroll
  for (int rr = 0; rr < 4; rr++) {
    int row = wv * 4 + rr;
    float s = 0.f, sq = 0.f;
#pragma unroll
    for (int k = 0; k < 8; k++) {
      float v = ldsX[row * 256 + l + k * 32];
      s += v; sq += v * v;
    }
    s = wred_sum(s); sq = wred_sum(sq);
    float mean = s * (1.f / 256.f);
    float var = sq * (1.f / 256.f) - mean * mean;
    float inv = rsqrtf(var + 1e-5f);
#pragma unroll
    for (int k = 0; k < 8; k++) {
      int c = l + k * 32;
      float v = (ldsX[row * 256 + c] - mean) * inv * gg[c] + bb[c];
      ldsH[row * 256 + c] = (_Float16)v;
    }
  }
  __syncthreads();
  { // coalesced store of h (f16) to global
    const uint4* src = (const uint4*)ldsH;
    uint4* dst = (uint4*)(Hout + row0 * WD);
    for (int e = threadIdx.x; e < 32 * 256 * 2 / 16; e += 256) dst[e] = src[e];
  }
  if (wv < 4) { // logits = h @ slice_w : 2x2 tiles of 16x16, K=256
    int mt = wv & 1, nt = wv >> 1;
    v8f acc = {};
#pragma unroll
    for (int ks = 0; ks < 8; ks++) {
      v16h a = frag_a(ldsH + mt * 16 * 256 + ks * 32, 256);
      v16h bf = frag_b(sliceWt + ((long)layer * MM + nt * 16) * WD + ks * 32, 256);
      acc = wmma16(a, bf, acc);
    }
#pragma unroll
    for (int r = 0; r < 8; r++) {
      int mm = mt * 16 + c_row(r);
      int nn = nt * 16 + c_col();
      ldsLg[mm * 32 + nn] = acc[r] + sliceB[layer * MM + nn];
    }
  }
  __syncthreads();
  float wacc = 0.f;
#pragma unroll
  for (int rr = 0; rr < 4; rr++) {
    int row = wv * 4 + rr;
    float v = ldsLg[row * 32 + l];
    float mx = wred_max(v);
    float e = __expf(v - mx);
    float sm = wred_sum(e);
    float w = e / sm;
    Wout[(row0 + row) * MM + l] = (_Float16)w;
    wacc += w;
  }
  atomicAdd(&ldsWS[l], wacc);
  __syncthreads();
  if (threadIdx.x < 32) atomicAdd(&WS[b * MM + threadIdx.x], ldsWS[threadIdx.x]);
}

// ---------------- z = w^T h (split-K, atomic reduce) ----------------
__global__ void __launch_bounds__(256) k_zmom(
    const _Float16* __restrict__ Hbuf, const _Float16* __restrict__ Wbuf,
    float* __restrict__ Z) {
  __shared__ _Float16 ldsWT[32 * 40];   // wT[m][k]
  __shared__ _Float16 ldsHT[256 * 40];  // hT[c][k]
  int b = blockIdx.y;
  long n0base = (long)b * NN + (long)blockIdx.x * 256;
  int wv = threadIdx.x >> 5;
  int mt = wv & 1, ngb = (wv >> 1) * 4;
  v8f acc[4] = {};
  for (int ch = 0; ch < 8; ch++) {
    __syncthreads();
    long n0 = n0base + ch * 32;
    for (int e = threadIdx.x; e < 32 * 32; e += 256) {
      int k = e >> 5, m = e & 31;
      ldsWT[m * 40 + k] = Wbuf[(n0 + k) * MM + m];
    }
    for (int e = threadIdx.x; e < 32 * 256; e += 256) {
      int k = e >> 8, c = e & 255;
      ldsHT[c * 40 + k] = Hbuf[(n0 + k) * WD + c];
    }
    __syncthreads();
    v16h a = frag_a(ldsWT + mt * 16 * 40, 40);
#pragma unroll
    for (int j = 0; j < 4; j++) {
      v16h bf = frag_b(ldsHT + (ngb + j) * 16 * 40, 40);
      acc[j] = wmma16(a, bf, acc[j]);
    }
  }
#pragma unroll
  for (int j = 0; j < 4; j++) {
    int nt = ngb + j;
#pragma unroll
    for (int r = 0; r < 8; r++) {
      int m = mt * 16 + c_row(r);
      int c = nt * 16 + c_col();
      atomicAdd(&Z[(b * MM + m) * WD + c], acc[j][r]);
    }
  }
}

// ---------------- qkv + attention + out projection (per batch) ----------------
__global__ void __launch_bounds__(256) k_attn(
    const float* __restrict__ Z, const float* __restrict__ WS,
    const _Float16* __restrict__ qkvWt, const float* __restrict__ qkvB,
    const _Float16* __restrict__ outWt, const float* __restrict__ outB,
    _Float16* __restrict__ ZPT, int layer) {
  __shared__ _Float16 bufA[32 * 256];   // z (f16), later o (f16)
  __shared__ _Float16 qkvh[32 * 768];
  int b = blockIdx.x;
  int wv = threadIdx.x >> 5, l = threadIdx.x & 31;
  for (int e = threadIdx.x; e < 32 * 256; e += 256) {
    int m = e >> 8;
    float ws = fmaxf(WS[b * MM + m], 1e-8f);
    bufA[e] = (_Float16)(Z[(long)b * MM * WD + e] / ws);
  }
  __syncthreads();
  { // qkv = z @ qkv_w + b : 2 x 48 tiles, K=256; hoist A frags (reused 12x)
    int mt = wv & 1, ngbase = (wv >> 1) * 12;
    v16h af[8];
#pragma unroll
    for (int ks = 0; ks < 8; ks++)
      af[ks] = frag_a(bufA + mt * 16 * 256 + ks * 32, 256);
    for (int j = 0; j < 12; j++) {
      int nt = ngbase + j;
      v8f acc = {};
#pragma unroll
      for (int ks = 0; ks < 8; ks++) {
        v16h bf = frag_b(qkvWt + ((long)layer * 768 + nt * 16) * WD + ks * 32, 256);
        acc = wmma16(af[ks], bf, acc);
      }
#pragma unroll
      for (int r = 0; r < 8; r++) {
        int m = mt * 16 + c_row(r);
        int c = nt * 16 + c_col();
        qkvh[m * 768 + c] = (_Float16)(acc[r] + qkvB[layer * 768 + c]);
      }
    }
  }
  __syncthreads();
  { // attention: one wave per head, lane = query row
    int h = wv, i = l;
    float q[32], s[32], o[32];
#pragma unroll
    for (int d = 0; d < 32; d++) q[d] = (float)qkvh[i * 768 + h * 32 + d];
#pragma unroll
    for (int j = 0; j < 32; j++) {
      float a = 0.f;
#pragma unroll
      for (int d = 0; d < 32; d++) a += q[d] * (float)qkvh[j * 768 + 256 + h * 32 + d];
      s[j] = a * 0.17677669529663687f;  // 1/sqrt(32)
    }
    float mx = -1e30f;
#pragma unroll
    for (int j = 0; j < 32; j++) mx = fmaxf(mx, s[j]);
    float sum = 0.f;
#pragma unroll
    for (int j = 0; j < 32; j++) { s[j] = __expf(s[j] - mx); sum += s[j]; }
    float inv = 1.f / sum;
#pragma unroll
    for (int d = 0; d < 32; d++) o[d] = 0.f;
#pragma unroll
    for (int j = 0; j < 32; j++) {
      float aj = s[j] * inv;
#pragma unroll
      for (int d = 0; d < 32; d++) o[d] += aj * (float)qkvh[j * 768 + 512 + h * 32 + d];
    }
    __syncthreads();  // all reads of bufA (z) done; reuse as o
#pragma unroll
    for (int d = 0; d < 32; d++) bufA[i * 256 + h * 32 + d] = (_Float16)o[d];
  }
  __syncthreads();
  { // zp = o @ out_w + out_b ; store transposed zp^T[c][m]; hoist A frags (4x reuse)
    int mt = wv & 1, ngb = (wv >> 1) * 4;
    v16h af[8];
#pragma unroll
    for (int ks = 0; ks < 8; ks++)
      af[ks] = frag_a(bufA + mt * 16 * 256 + ks * 32, 256);
    for (int j = 0; j < 4; j++) {
      int nt = ngb + j;
      v8f acc = {};
#pragma unroll
      for (int ks = 0; ks < 8; ks++) {
        v16h bf = frag_b(outWt + ((long)layer * WD + nt * 16) * WD + ks * 32, 256);
        acc = wmma16(af[ks], bf, acc);
      }
#pragma unroll
      for (int r = 0; r < 8; r++) {
        int m = mt * 16 + c_row(r);
        int c = nt * 16 + c_col();
        float zp = acc[r] + outB[layer * WD + c];
        ZPT[(long)b * WD * MM + c * MM + m] = (_Float16)zp;
      }
    }
  }
}

// ---------------- broadcast-back + residual + LN2 + FFN (fused) ----------------
__global__ void __launch_bounds__(256) k_ffn(
    float* __restrict__ X, const _Float16* __restrict__ Wbuf,
    const _Float16* __restrict__ ZPT, const float* __restrict__ ln2g,
    const float* __restrict__ ln2b, const _Float16* __restrict__ ffn1Wt,
    const float* __restrict__ ffn1B, const _Float16* __restrict__ ffn2Wt,
    const float* __restrict__ ffn2B, int layer) {
  __shared__ float ldsX[32 * 256];      // 32KB
  __shared__ _Float16 ldsH2[32 * 256];  // 16KB
  __shared__ _Float16 ldsY1[32 * 256];  // 16KB
  long row0 = (long)blockIdx.x * 32;
  int b = (int)(row0 / NN);
  int wv = threadIdx.x >> 5, l = threadIdx.x & 31;
  const _Float16* w1base = ffn1Wt + (long)layer * FFD * WD;
  const _Float16* w2base = ffn2Wt + (long)layer * WD * FFD;
  // warm L2/WGP$ for this layer's FFN weights (256 thr x 2KB = 512KB each)
  __builtin_prefetch((const char*)w1base + threadIdx.x * 2048, 0, 3);
  __builtin_prefetch((const char*)w2base + threadIdx.x * 2048, 0, 3);
  for (int e = threadIdx.x; e < 32 * 256; e += 256) ldsX[e] = X[row0 * WD + e];
  __syncthreads();
  int mt = wv & 1, ngb = (wv >> 1) * 4;
  { // x += w @ zp   (K = M = 32, one WMMA per tile); operands straight from global (L2-hot)
    v16h a = frag_a(Wbuf + (row0 + mt * 16) * MM, 32);
#pragma unroll
    for (int j = 0; j < 4; j++) {
      int nt = ngb + j;
      v16h bf = frag_b(ZPT + (long)b * WD * MM + nt * 16 * MM, 32);
      v8f acc = {};
      acc = wmma16(a, bf, acc);
#pragma unroll
      for (int r = 0; r < 8; r++) {
        int m = mt * 16 + c_row(r);
        int c = nt * 16 + c_col();
        ldsX[m * 256 + c] += acc[r];
      }
    }
  }
  __syncthreads();
  const float* g2 = ln2g + layer * WD;
  const float* b2 = ln2b + layer * WD;
#pragma unroll
  for (int rr = 0; rr < 4; rr++) {
    int row = wv * 4 + rr;
    float s = 0.f, sq = 0.f;
#pragma unroll
    for (int k = 0; k < 8; k++) {
      float v = ldsX[row * 256 + l + k * 32];
      s += v; sq += v * v;
    }
    s = wred_sum(s); sq = wred_sum(sq);
    float mean = s * (1.f / 256.f);
    float var = sq * (1.f / 256.f) - mean * mean;
    float inv = rsqrtf(var + 1e-5f);
#pragma unroll
    for (int k = 0; k < 8; k++) {
      int c = l + k * 32;
      float v = (ldsX[row * 256 + c] - mean) * inv * g2[c] + b2[c];
      ldsH2[row * 256 + c] = (_Float16)v;
    }
  }
  __syncthreads();
  // Hoist GEMM1 A fragments once: ldsH2 is invariant for the whole FFN section.
  v16h a1[8];
#pragma unroll
  for (int ks = 0; ks < 8; ks++)
    a1[ks] = frag_a(ldsH2 + mt * 16 * 256 + ks * 32, 256);
  v8f c2[4] = {};
  for (int fc = 0; fc < 4; fc++) {  // FF processed in 256-wide chunks
    // GEMM1 chunk: y1 = gelu(h2 @ ffn1_w + b1)
    for (int j = 0; j < 4; j++) {
      int nt = ngb + j;
      v8f acc = {};
#pragma unroll
      for (int ks = 0; ks < 8; ks++) {
        v16h bf = frag_b(w1base + ((long)fc * 256 + nt * 16) * WD + ks * 32, 256);
        acc = wmma16(a1[ks], bf, acc);
      }
#pragma unroll
      for (int r = 0; r < 8; r++) {
        int m = mt * 16 + c_row(r);
        int c = nt * 16 + c_col();
        float v = acc[r] + ffn1B[layer * FFD + fc * 256 + c];
        v = 0.5f * v * (1.f + erff(v * 0.70710678118654752f));  // exact gelu
        ldsY1[m * 256 + c] = (_Float16)v;
      }
    }
    __syncthreads();
    // GEMM2 partial: c2 += y1 @ ffn2_w[chunk]; hoist A frags (4x reuse per chunk)
    v16h a2[8];
#pragma unroll
    for (int ks = 0; ks < 8; ks++)
      a2[ks] = frag_a(ldsY1 + mt * 16 * 256 + ks * 32, 256);
    for (int j = 0; j < 4; j++) {
      int nt = ngb + j;
#pragma unroll
      for (int ks = 0; ks < 8; ks++) {
        v16h bf = frag_b(w2base + (long)(nt * 16) * FFD + fc * 256 + ks * 32, 1024);
        c2[j] = wmma16(a2[ks], bf, c2[j]);
      }
    }
    __syncthreads();
  }
#pragma unroll
  for (int j = 0; j < 4; j++) {
    int nt = ngb + j;
#pragma unroll
    for (int r = 0; r < 8; r++) {
      int m = mt * 16 + c_row(r);
      int c = nt * 16 + c_col();
      X[(row0 + m) * WD + c] = ldsX[m * 256 + c] + c2[j][r] + ffn2B[layer * WD + c];
    }
  }
}

// ---------------- final projection ----------------
__global__ void __launch_bounds__(256) k_proj(
    const float* __restrict__ X, const _Float16* __restrict__ projWt,
    const float* __restrict__ projB, float* __restrict__ out) {
  __shared__ _Float16 ldsA[32 * 256];
  long row0 = (long)blockIdx.x * 32;
  for (int e = threadIdx.x; e < 32 * 256; e += 256)
    ldsA[e] = (_Float16)X[row0 * WD + e];
  __syncthreads();
  int wv = threadIdx.x >> 5;
  if (wv < 2) {
    int mt = wv;
    v8f acc = {};
#pragma unroll
    for (int ks = 0; ks < 8; ks++) {
      v16h a = frag_a(ldsA + mt * 16 * 256 + ks * 32, 256);
      v16h bf = frag_b(projWt + ks * 32, 256);
      acc = wmma16(a, bf, acc);
    }
#pragma unroll
    for (int r = 0; r < 8; r++) {
      int m = mt * 16 + c_row(r);
      int c = c_col();
      if (c < CIN_) out[(row0 + m) * CIN_ + c] = acc[r] + projB[c];
    }
  }
}

// ---------------- host ----------------
extern "C" void kernel_launch(void* const* d_in, const int* in_sizes, int n_in,
                              void* d_out, int out_size, void* d_ws, size_t ws_size,
                              hipStream_t stream) {
  (void)in_sizes; (void)n_in; (void)out_size; (void)ws_size;
  const float* feat    = (const float*)d_in[0];
  const float* coords  = (const float*)d_in[1];
  const float* tnorm   = (const float*)d_in[2];
  const float* embed_w = (const float*)d_in[3];
  const float* embed_b = (const float*)d_in[4];
  const float* ln1g    = (const float*)d_in[5];
  const float* ln1b    = (const float*)d_in[6];
  const float* slice_w = (const float*)d_in[7];
  const float* slice_b = (const float*)d_in[8];
  const float* qkv_w   = (const float*)d_in[9];
  const float* qkv_b   = (const float*)d_in[10];
  const float* out_w   = (const float*)d_in[11];
  const float* out_b   = (const float*)d_in[12];
  const float* ln2g    = (const float*)d_in[13];
  const float* ln2b    = (const float*)d_in[14];
  const float* ffn1_w  = (const float*)d_in[15];
  const float* ffn1_b  = (const float*)d_in[16];
  const float* ffn2_w  = (const float*)d_in[17];
  const float* ffn2_b  = (const float*)d_in[18];
  const float* proj_w  = (const float*)d_in[19];
  const float* proj_b  = (const float*)d_in[20];
  float* out = (float*)d_out;

  char* p = (char*)d_ws;
  auto take = [&](size_t bytes) -> char* {
    char* r = p;
    p += (bytes + 255) & ~(size_t)255;
    return r;
  };
  float*    X       = (float*)take((size_t)BB * NN * WD * 4);
  _Float16* Hbuf    = (_Float16*)take((size_t)BB * NN * WD * 2);
  _Float16* Wbuf    = (_Float16*)take((size_t)BB * NN * MM * 2);
  float*    WS      = (float*)take((size_t)BB * MM * 4);
  float*    Zbuf    = (float*)take((size_t)BB * MM * WD * 4);
  _Float16* ZPT     = (_Float16*)take((size_t)BB * WD * MM * 2);
  _Float16* embedWt = (_Float16*)take((size_t)WD * 32 * 2);
  _Float16* sliceWt = (_Float16*)take((size_t)LL * MM * WD * 2);
  _Float16* qkvWt   = (_Float16*)take((size_t)LL * 3 * WD * WD * 2);
  _Float16* outWt   = (_Float16*)take((size_t)LL * WD * WD * 2);
  _Float16* ffn1Wt  = (_Float16*)take((size_t)LL * FFD * WD * 2);
  _Float16* ffn2Wt  = (_Float16*)take((size_t)LL * WD * FFD * 2);
  _Float16* projWt  = (_Float16*)take((size_t)16 * WD * 2);

  // weight prep (f32 -> f16, transposed to [col][k])
  k_prep_embed<<<32, 256, 0, stream>>>(embed_w, embedWt);
  k_prep_proj<<<16, 256, 0, stream>>>(proj_w, projWt);
  auto tl = [&](const float* s, _Float16* d, int R, int C) {
    long tot = (long)LL * R * C;
    k_transpose_l<<<(int)((tot + 255) / 256), 256, 0, stream>>>(s, d, R, C, LL);
  };
  tl(slice_w, sliceWt, WD, MM);
  tl(qkv_w, qkvWt, WD, 3 * WD);
  tl(out_w, outWt, WD, WD);
  tl(ffn1_w, ffn1Wt, WD, FFD);
  tl(ffn2_w, ffn2Wt, FFD, WD);

  const int nblk = BB * NN / 32;  // 2048
  k_embed<<<nblk, 256, 0, stream>>>(feat, coords, tnorm, embedWt, embed_b, X);
  for (int l = 0; l < LL; l++) {
    k_zero<<<(BB * MM * WD + 255) / 256, 256, 0, stream>>>(Zbuf, WS);
    k_slice<<<nblk, 256, 0, stream>>>(X, ln1g, ln1b, sliceWt, slice_b, Hbuf, Wbuf, WS, l);
    k_zmom<<<dim3(NN / 256, BB), 256, 0, stream>>>(Hbuf, Wbuf, Zbuf);
    k_attn<<<BB, 256, 0, stream>>>(Zbuf, WS, qkvWt, qkv_b, outWt, out_b, ZPT, l);
    k_ffn<<<nblk, 256, 0, stream>>>(X, Wbuf, ZPT, ln2g, ln2b, ffn1Wt, ffn1_b,
                                    ffn2Wt, ffn2_b, l);
  }
  k_proj<<<nblk, 256, 0, stream>>>(X, projWt, proj_b, out);
}
